// PrefillModel_18322330485526
// MI455X (gfx1250) — matrix operand order repaired and checked by reading, not compiled, not run on hardware
//
#include <hip/hip_runtime.h>
#include <math.h>

#define SQ    1024      // sequence length
#define EMB   1024      // model dim
#define NH    16        // heads
#define HDm   64        // head dim
#define NL    12        // layers
#define VOC   50257
#define INNER 4096

typedef __attribute__((ext_vector_type(16))) __bf16 v16bf;
typedef __attribute__((ext_vector_type(8)))  __bf16 v8bf;
typedef __attribute__((ext_vector_type(8)))  float  v8f;

__device__ __forceinline__ unsigned short f2bf(float f) {
    unsigned int u = __float_as_uint(f);
    u += 0x7fffu + ((u >> 16) & 1u);            // round-to-nearest-even
    return (unsigned short)(u >> 16);
}

// pack two fp32 -> two bf16 in one DWORD (lo in [15:0], hi in [31:16])
__device__ __forceinline__ unsigned int f2bf2(float lo, float hi) {
    return ((unsigned int)f2bf(hi) << 16) | (unsigned int)f2bf(lo);
}

__device__ __forceinline__ v16bf cat8(v8bf lo, v8bf hi) {
    return __builtin_shufflevector(lo, hi, 0,1,2,3,4,5,6,7,8,9,10,11,12,13,14,15);
}

__device__ __forceinline__ float gelu_new(float x) {
    const float c = 0.7978845608028654f;
    float t = tanhf(c * (x + 0.044715f * x * x * x));
    return 0.5f * x * (1.0f + t);
}

// ---------------------------------------------------------------- embedding
__global__ __launch_bounds__(256) void embed_kernel(const int* __restrict__ ids,
        const float* __restrict__ wte, const float* __restrict__ wpe,
        float* __restrict__ h) {
    int s = blockIdx.x;
    long tok = ids[s];
    for (int i = 0; i < EMB / 256; ++i) {
        int e = threadIdx.x + i * 256;
        h[(long)s * EMB + e] = wte[tok * EMB + e] + wpe[(long)s * EMB + e];
    }
}

// ---------------------------------------------------------------- layernorm
__global__ __launch_bounds__(256) void layernorm_kernel(const float* __restrict__ in,
        const float* __restrict__ g, const float* __restrict__ b,
        float* __restrict__ out) {
    __shared__ float r1[256], r2[256];
    int s = blockIdx.x;
    const float* row = in + (long)s * EMB;
    float v[4];
    float sum = 0.f, sq = 0.f;
    for (int i = 0; i < 4; ++i) {
        v[i] = row[threadIdx.x + i * 256];
        sum += v[i]; sq += v[i] * v[i];
    }
    r1[threadIdx.x] = sum; r2[threadIdx.x] = sq;
    __syncthreads();
    for (int off = 128; off > 0; off >>= 1) {
        if (threadIdx.x < off) {
            r1[threadIdx.x] += r1[threadIdx.x + off];
            r2[threadIdx.x] += r2[threadIdx.x + off];
        }
        __syncthreads();
    }
    float mu  = r1[0] * (1.0f / EMB);
    float var = r2[0] * (1.0f / EMB) - mu * mu;
    float inv = rsqrtf(var + 1e-5f);
    for (int i = 0; i < 4; ++i) {
        int e = threadIdx.x + i * 256;
        out[(long)s * EMB + e] = (v[i] - mu) * inv * g[e] + b[e];
    }
}

// ---------------------------------------------------------------- WMMA GEMM
// C[M,N] = A[M,K] * op(B) + bias (+resid) (+gelu)
// BT=true : B is [N,K] row-major (C = A @ B^T : x@W.T pattern and Q@K^T)
// BT=false: B is [K,N] row-major (C = A @ B   : P@V pattern)
// Macro tile 128(M) x 64(N) x 32(K); 8 waves, each owning a 32x32 output
// (2x2 WMMA tiles -> 4 v_wmma per K-step). Next K-chunk is prefetched from
// global (float2 loads) into registers while the current chunk's WMMAs run;
// staging converts fp32 -> packed 2xbf16 DWORD stores into LDS.
// blockIdx.z batches over heads via aBatch/bBatch/cBatch element strides.
template<bool BT>
__global__ __launch_bounds__(256) void gemm_wmma_kernel(
        const float* __restrict__ A, const float* __restrict__ B,
        const float* __restrict__ bias, const float* resid, float* C,
        int M, int N, int K, int lda, int ldb, int ldc,
        long aBatch, long bBatch, long cBatch, int act) {
    constexpr int LSTR = 40;                    // 80B row stride: 16B aligned, conflict-spread
    __shared__ unsigned short As[128 * LSTR];   // 128 x 32 bf16 A tile
    __shared__ unsigned short Bs[64 * LSTR];    // 64(N) x 32(K) bf16, always [n][k]
    int z = blockIdx.z;
    A += z * aBatch; B += z * bBatch; C += z * cBatch;
    if (resid) resid += z * cBatch;
    int bm = blockIdx.y * 128, bn = blockIdx.x * 64;
    int tid = threadIdx.x, lane = tid & 31;
    int wid = tid >> 5;
    int waveM = wid >> 1;                       // 0..3 -> 32 rows each
    int waveN = wid & 1;                        // 0..1 -> 32 cols each
    int mloc = lane & 15, ksel = lane >> 4;

    v8f acc[2][2];
    #pragma unroll
    for (int i = 0; i < 2; ++i)
        #pragma unroll
        for (int j = 0; j < 2; ++j)
            acc[i][j] = (v8f){0,0,0,0,0,0,0,0};

    float2 aReg[8], bReg[4];
    auto loadA = [&](int k0) {                  // 128x32, K-pairs per thread
        #pragma unroll
        for (int i = 0; i < 8; ++i) {
            int e = tid + i * 256;              // 0..2047 pair index
            int r = e >> 4, c = (e & 15) * 2;
            aReg[i] = *(const float2*)&A[(long)(bm + r) * lda + k0 + c];
        }
    };
    auto loadB = [&](int k0) {
        if (BT) {                               // 64x32, K-pairs per thread
            #pragma unroll
            for (int i = 0; i < 4; ++i) {
                int e = tid + i * 256;          // 0..1023 pair index
                int r = e >> 4, c = (e & 15) * 2;
                bReg[i] = *(const float2*)&B[(long)(bn + r) * ldb + k0 + c];
            }
        } else {                                // [K,N] source: N-pairs per thread
            #pragma unroll
            for (int i = 0; i < 4; ++i) {
                int e = tid + i * 256;
                int k = e >> 5, n = (e & 31) * 2;
                bReg[i] = *(const float2*)&B[(long)(k0 + k) * ldb + bn + n];
            }
        }
    };

    loadA(0); loadB(0);

    for (int k0 = 0; k0 < K; k0 += 32) {
        // commit current chunk (regs -> LDS, fp32 -> packed bf16 pairs)
        #pragma unroll
        for (int i = 0; i < 8; ++i) {
            int e = tid + i * 256;
            int r = e >> 4, c = (e & 15) * 2;
            *(unsigned int*)&As[r * LSTR + c] = f2bf2(aReg[i].x, aReg[i].y);
        }
        if (BT) {
            #pragma unroll
            for (int i = 0; i < 4; ++i) {
                int e = tid + i * 256;
                int r = e >> 4, c = (e & 15) * 2;
                *(unsigned int*)&Bs[r * LSTR + c] = f2bf2(bReg[i].x, bReg[i].y);
            }
        } else {
            #pragma unroll
            for (int i = 0; i < 4; ++i) {
                int e = tid + i * 256;
                int k = e >> 5, n = (e & 31) * 2;
                Bs[n * LSTR + k]       = f2bf(bReg[i].x);
                Bs[(n + 1) * LSTR + k] = f2bf(bReg[i].y);
            }
        }
        __syncthreads();

        // prefetch next chunk from global while WMMAs consume LDS
        int kn = k0 + 32;
        if (kn < K) { loadA(kn); loadB(kn); }

        // B fragments: lane holds col n, K = ksel*16 .. ksel*16+15 (contiguous)
        const unsigned short* b0 = &Bs[(waveN * 32 + mloc) * LSTR + ksel * 16];
        const unsigned short* b1 = &Bs[(waveN * 32 + 16 + mloc) * LSTR + ksel * 16];
        v16bf bf0 = cat8(*(const v8bf*)b0, *(const v8bf*)(b0 + 8));
        v16bf bf1 = cat8(*(const v8bf*)b1, *(const v8bf*)(b1 + 8));

        // A fragments: lane holds row m, K = ksel*16 .. ksel*16+15 (contiguous)
        const unsigned short* a0 = &As[(waveM * 32 + mloc) * LSTR];
        const unsigned short* a1 = &As[(waveM * 32 + 16 + mloc) * LSTR];
        v16bf af0 = cat8(*(const v8bf*)(a0 + ksel * 16), *(const v8bf*)(a0 + ksel * 16 + 8));
        v16bf af1 = cat8(*(const v8bf*)(a1 + ksel * 16), *(const v8bf*)(a1 + ksel * 16 + 8));

        acc[0][0] = __builtin_amdgcn_wmma_f32_16x16x32_bf16(false, af0, false, bf0,
                                                            (short)0, acc[0][0], false, false);
        acc[0][1] = __builtin_amdgcn_wmma_f32_16x16x32_bf16(false, af0, false, bf1,
                                                            (short)0, acc[0][1], false, false);
        acc[1][0] = __builtin_amdgcn_wmma_f32_16x16x32_bf16(false, af1, false, bf0,
                                                            (short)0, acc[1][0], false, false);
        acc[1][1] = __builtin_amdgcn_wmma_f32_16x16x32_bf16(false, af1, false, bf1,
                                                            (short)0, acc[1][1], false, false);
        __syncthreads();
    }

    // epilogue: D layout -> m = r + 8*(lane>>4), n = lane&15
    int nl = lane & 15, half = lane >> 4;
    float bv[2];
    int gn[2];
    #pragma unroll
    for (int tn = 0; tn < 2; ++tn) {
        gn[tn] = bn + waveN * 32 + tn * 16 + nl;
        bv[tn] = bias ? bias[gn[tn]] : 0.0f;
    }
    #pragma unroll
    for (int tm = 0; tm < 2; ++tm) {
        int mb = bm + waveM * 32 + tm * 16 + half * 8;
        #pragma unroll
        for (int tn = 0; tn < 2; ++tn) {
            #pragma unroll
            for (int r = 0; r < 8; ++r) {
                float v = acc[tm][tn][r] + bv[tn];
                if (act == 1) v = gelu_new(v);
                long ci = (long)(mb + r) * ldc + gn[tn];
                if (resid) v += resid[ci];
                C[ci] = v;
            }
        }
    }
}

// ------------------------------------------- scale + bias + causal softmax
__global__ __launch_bounds__(256) void attn_softmax_kernel(
        float* scores, const float* __restrict__ ab) {
    __shared__ float red[256];
    int q = blockIdx.x, hh = blockIdx.y;
    float* row = scores + ((long)hh * SQ + q) * SQ;
    const float* brow = ab + (long)q * SQ;
    float v[4];
    float mx = -1e30f;
    for (int i = 0; i < 4; ++i) {
        int k = threadIdx.x + i * 256;
        float val = (k <= q) ? row[k] * 0.125f + brow[k] : -1e9f;
        v[i] = val;
        mx = fmaxf(mx, val);
    }
    red[threadIdx.x] = mx; __syncthreads();
    for (int off = 128; off > 0; off >>= 1) {
        if (threadIdx.x < off) red[threadIdx.x] = fmaxf(red[threadIdx.x], red[threadIdx.x + off]);
        __syncthreads();
    }
    mx = red[0]; __syncthreads();
    float sum = 0.f;
    for (int i = 0; i < 4; ++i) { v[i] = expf(v[i] - mx); sum += v[i]; }
    red[threadIdx.x] = sum; __syncthreads();
    for (int off = 128; off > 0; off >>= 1) {
        if (threadIdx.x < off) red[threadIdx.x] += red[threadIdx.x + off];
        __syncthreads();
    }
    float inv = 1.0f / red[0];
    for (int i = 0; i < 4; ++i) {
        int k = threadIdx.x + i * 256;
        row[k] = v[i] * inv;
    }
}

// ---------------------------------------------------------------- KV cache
__global__ __launch_bounds__(256) void kvcopy_kernel(const float* __restrict__ qkv,
        float* __restrict__ kc, float* __restrict__ vc, int l) {
    int s = blockIdx.x;
    for (int i = 0; i < 4; ++i) {
        int idx = threadIdx.x + i * 256;
        int hh = idx >> 6, d = idx & 63;
        long src = (long)s * 3 * EMB + hh * HDm + d;
        long dst = (((long)l * NH + hh) * SQ + s) * HDm + d;
        kc[dst] = qkv[src + EMB];
        vc[dst] = qkv[src + 2 * EMB];
    }
}

// ---------------------------------------------------- last-token lm_head
__global__ __launch_bounds__(256) void logits_kernel(const float* __restrict__ hlast,
        const float* __restrict__ wte, float* __restrict__ logits) {
    int lane = threadIdx.x & 31, wid = threadIdx.x >> 5;
    long row = (long)blockIdx.x * 8 + wid;
    if (row >= VOC) return;
    const float* w = wte + row * EMB;
    float sum = 0.f;
    for (int j = 0; j < EMB / 32; ++j)
        sum += hlast[lane + j * 32] * w[lane + j * 32];
    for (int off = 16; off > 0; off >>= 1)
        sum += __shfl_down(sum, off, 32);
    if (lane == 0) logits[row] = sum;
}

// -------------------------------------------------- vocab softmax + argmax
__global__ __launch_bounds__(1024) void vocab_softmax_kernel(
        const float* __restrict__ logits, float* __restrict__ out) {
    __shared__ float smax[1024];
    __shared__ int   sidx[1024];
    __shared__ float ssum[1024];
    int t = threadIdx.x;
    float mx = -1e30f; int mi = 0;
    for (int i = t; i < VOC; i += 1024) {
        float v = logits[i];
        if (v > mx) { mx = v; mi = i; }
    }
    smax[t] = mx; sidx[t] = mi; __syncthreads();
    for (int off = 512; off > 0; off >>= 1) {
        if (t < off) {
            if (smax[t + off] > smax[t] ||
                (smax[t + off] == smax[t] && sidx[t + off] < sidx[t])) {
                smax[t] = smax[t + off]; sidx[t] = sidx[t + off];
            }
        }
        __syncthreads();
    }
    float gmx = smax[0];
    float sum = 0.f;
    for (int i = t; i < VOC; i += 1024) sum += expf(logits[i] - gmx);
    ssum[t] = sum; __syncthreads();
    for (int off = 512; off > 0; off >>= 1) {
        if (t < off) ssum[t] += ssum[t + off];
        __syncthreads();
    }
    float inv = 1.0f / ssum[0];
    for (int i = t; i < VOC; i += 1024)
        out[1 + i] = expf(logits[i] - gmx) * inv;
    if (t == 0) out[0] = (float)sidx[0];
}

// ---------------------------------------------------------------- launcher
extern "C" void kernel_launch(void* const* d_in, const int* in_sizes, int n_in,
                              void* d_out, int out_size, void* d_ws, size_t ws_size,
                              hipStream_t stream) {
    (void)in_sizes; (void)n_in; (void)out_size; (void)ws_size;
    const int*   ids   = (const int*)d_in[0];
    const float* wte   = (const float*)d_in[1];
    const float* wpe   = (const float*)d_in[2];
    const float* ln1g  = (const float*)d_in[3];
    const float* ln1b  = (const float*)d_in[4];
    const float* caw   = (const float*)d_in[5];
    const float* cab   = (const float*)d_in[6];
    const float* abias = (const float*)d_in[7];
    const float* apw   = (const float*)d_in[8];
    const float* apb   = (const float*)d_in[9];
    const float* ln2g  = (const float*)d_in[10];
    const float* ln2b  = (const float*)d_in[11];
    const float* fw    = (const float*)d_in[12];
    const float* fb    = (const float*)d_in[13];
    const float* pw    = (const float*)d_in[14];
    const float* pb    = (const float*)d_in[15];
    const float* lnfg  = (const float*)d_in[16];
    const float* lnfb  = (const float*)d_in[17];

    float* ws      = (float*)d_ws;
    float* h       = ws;                               // S*E
    float* x       = h       + (long)SQ * EMB;         // S*E
    float* qkv     = x       + (long)SQ * EMB;         // S*3E
    float* scores  = qkv     + (long)SQ * 3 * EMB;     // H*S*S
    float* attnout = scores  + (long)NH * SQ * SQ;     // S*E
    float* mbuf    = attnout + (long)SQ * EMB;         // S*I
    float* logits  = mbuf    + (long)SQ * INNER;       // V

    float* out = (float*)d_out;
    float* kc  = out + 1 + VOC;
    float* vc  = kc + (long)NL * NH * SQ * HDm;

    dim3 blk(256);
    embed_kernel<<<SQ, blk, 0, stream>>>(ids, wte, wpe, h);

    for (int l = 0; l < NL; ++l) {
        layernorm_kernel<<<SQ, blk, 0, stream>>>(h, ln1g + (long)l * EMB,
                                                 ln1b + (long)l * EMB, x);
        // qkv = x @ c_attn_w^T + b
        gemm_wmma_kernel<true><<<dim3(3 * EMB / 64, SQ / 128, 1), blk, 0, stream>>>(
            x, caw + (long)l * 3 * EMB * EMB, cab + (long)l * 3 * EMB, nullptr, qkv,
            SQ, 3 * EMB, EMB, EMB, EMB, 3 * EMB, 0, 0, 0, 0);
        kvcopy_kernel<<<SQ, blk, 0, stream>>>(qkv, kc, vc, l);
        // scores[h] = Q @ K^T  (per-head via z batch; head stride 64 cols in qkv)
        gemm_wmma_kernel<true><<<dim3(SQ / 64, SQ / 128, NH), blk, 0, stream>>>(
            qkv, qkv + EMB, nullptr, nullptr, scores,
            SQ, SQ, HDm, 3 * EMB, 3 * EMB, SQ, HDm, HDm, (long)SQ * SQ, 0);
        attn_softmax_kernel<<<dim3(SQ, NH), blk, 0, stream>>>(scores,
                                                              abias + (long)l * SQ * SQ);
        // ctx[h] = P @ V
        gemm_wmma_kernel<false><<<dim3(HDm / 64, SQ / 128, NH), blk, 0, stream>>>(
            scores, qkv + 2 * EMB, nullptr, nullptr, attnout,
            SQ, HDm, SQ, SQ, 3 * EMB, EMB, (long)SQ * SQ, HDm, HDm, 0);
        // h += ctx @ atproj_w^T + b
        gemm_wmma_kernel<true><<<dim3(EMB / 64, SQ / 128, 1), blk, 0, stream>>>(
            attnout, apw + (long)l * EMB * EMB, apb + (long)l * EMB, h, h,
            SQ, EMB, EMB, EMB, EMB, EMB, 0, 0, 0, 0);
        layernorm_kernel<<<SQ, blk, 0, stream>>>(h, ln2g + (long)l * EMB,
                                                 ln2b + (long)l * EMB, x);
        // m = gelu(x @ cfc_w^T + b)
        gemm_wmma_kernel<true><<<dim3(INNER / 64, SQ / 128, 1), blk, 0, stream>>>(
            x, fw + (long)l * INNER * EMB, fb + (long)l * INNER, nullptr, mbuf,
            SQ, INNER, EMB, EMB, EMB, INNER, 0, 0, 0, 1);
        // h += m @ mproj_w^T + b
        gemm_wmma_kernel<true><<<dim3(EMB / 64, SQ / 128, 1), blk, 0, stream>>>(
            mbuf, pw + (long)l * EMB * INNER, pb + (long)l * EMB, h, h,
            SQ, EMB, INNER, INNER, INNER, EMB, 0, 0, 0, 0);
    }

    layernorm_kernel<<<SQ, blk, 0, stream>>>(h, lnfg, lnfb, x);
    logits_kernel<<<(VOC + 7) / 8, blk, 0, stream>>>(x + (long)(SQ - 1) * EMB, wte, logits);
    vocab_softmax_kernel<<<1, 1024, 0, stream>>>(logits, out);
}
